// MultiHeadAttention_35914516529855
// MI455X (gfx1250) — matrix operand order repaired and checked by reading, not compiled, not run on hardware
//
#include <hip/hip_runtime.h>

// ---------------------------------------------------------------------------
// MultiHeadAttention forward for MI455X (gfx1250), bf16 WMMA pipeline.
//   B=4, S=2048, D=1024, H=16, DK=64
// Stages:
//   0) fp32 -> bf16 pre-conversion of weights/activations (if ws allows)
//   1) Q/K/V projections as bf16 WMMA GEMMs (V written transposed)
//   2) flash attention with TRANSPOSED score tiles (S^T = K x Q^T):
//      softmax axis lives in VGPRs -> register reductions, and the S^T
//      D-fragment is bit-identical to the P.V A-fragment -> no LDS transpose.
//   3) output projection WMMA GEMM -> fp32
// ---------------------------------------------------------------------------

#define B_DIM 4
#define S_DIM 2048
#define D_DIM 1024
#define H_DIM 16
#define DK_DIM 64
#define NEG_INF (-1e12f)

typedef __attribute__((ext_vector_type(16))) __bf16 bf16x16;
typedef __attribute__((ext_vector_type(8)))  float  floatx8;

// ---- WMMA fragment helpers (layouts per CDNA5 ISA 7.12.2) ------------------
// A matrix 16x32 bf16: lane = 16*half + m; slots j<8 -> K = half*8+j,
//                      slots j>=8 -> K = 16 + half*8 + (j-8).
__device__ __forceinline__ bf16x16 frag_a_f32(const float* __restrict__ row, int half) {
  bf16x16 a;
#pragma unroll
  for (int j = 0; j < 8; ++j) a[j] = (__bf16)row[half * 8 + j];
#pragma unroll
  for (int j = 0; j < 8; ++j) a[8 + j] = (__bf16)row[16 + half * 8 + j];
  return a;
}
__device__ __forceinline__ bf16x16 frag_a_bf16(const __bf16* __restrict__ row, int half) {
  bf16x16 a;
#pragma unroll
  for (int j = 0; j < 8; ++j) a[j] = row[half * 8 + j];
#pragma unroll
  for (int j = 0; j < 8; ++j) a[8 + j] = row[16 + half * 8 + j];
  return a;
}
// B matrix 32x16 bf16: lane = 16*half + n; slots j<8 -> K = half*8+j,
//                      slots j>=8 -> K = 16 + half*8 + (j-8).
// Source is a row of a K-major (row-major [N][K]) matrix -> contiguous.
__device__ __forceinline__ bf16x16 frag_b_f32(const float* __restrict__ row, int half) {
  bf16x16 b;
#pragma unroll
  for (int j = 0; j < 8; ++j) b[j] = (__bf16)row[half * 8 + j];
#pragma unroll
  for (int j = 0; j < 8; ++j) b[8 + j] = (__bf16)row[16 + half * 8 + j];
  return b;
}
__device__ __forceinline__ bf16x16 frag_b_bf16(const __bf16* __restrict__ row, int half) {
  bf16x16 b;
#pragma unroll
  for (int j = 0; j < 8; ++j) b[j] = row[half * 8 + j];
#pragma unroll
  for (int j = 0; j < 8; ++j) b[8 + j] = row[16 + half * 8 + j];
  return b;
}

__device__ __forceinline__ floatx8 wmma_bf16(bf16x16 a, bf16x16 b, floatx8 c) {
  return __builtin_amdgcn_wmma_f32_16x16x32_bf16(
      /*neg_a=*/false, a, /*neg_b=*/false, b,
      /*c_mod=*/(short)0, c, /*reuse_a=*/false, /*reuse_b=*/false);
}

// ---------------------------------------------------------------------------
// fp32 -> bf16 conversion (grid-stride, float4 vectorized)
// ---------------------------------------------------------------------------
__global__ __launch_bounds__(256) void cvt_f32_to_bf16(
    const float* __restrict__ src, __bf16* __restrict__ dst, int n4) {
  const int stride = gridDim.x * blockDim.x;
  for (int i = blockIdx.x * blockDim.x + threadIdx.x; i < n4; i += stride) {
    const float4 v = ((const float4*)src)[i];
    __bf16 o0 = (__bf16)v.x, o1 = (__bf16)v.y, o2 = (__bf16)v.z, o3 = (__bf16)v.w;
    dst[4 * i + 0] = o0; dst[4 * i + 1] = o1;
    dst[4 * i + 2] = o2; dst[4 * i + 3] = o3;
  }
}

// ---------------------------------------------------------------------------
// GEMM:  C[M=8192, N=1024] = A[M,K=1024] * W[N,K]^T  (i.e. x @ W.T)
// Wave tile 64x64 (4x4 WMMA), WG 8 waves = 256x128 tile.
// OUT_MODE 0: bf16 [B,H,S,DK]   (Q, K)
// OUT_MODE 1: bf16 [B,H,DK,S]   (V transposed for PV WMMA)
// OUT_MODE 2: f32  [B*S, D]     (final output)
// ---------------------------------------------------------------------------
template <int OUT_MODE, bool A_BF, bool W_BF>
__global__ __launch_bounds__(256) void gemm_wmma_bf16(
    const float* __restrict__ Af, const __bf16* __restrict__ Ab,
    const float* __restrict__ Wf, const __bf16* __restrict__ Wb,
    void* __restrict__ outp) {
  const int lane = threadIdx.x & 31;
  const int wave = threadIdx.x >> 5;
  const int half = lane >> 4;
  const int ln   = lane & 15;
  const int wm = wave & 3;          // 0..3
  const int wn = wave >> 2;         // 0..1
  const int m0 = blockIdx.x * 256 + wm * 64;
  const int n0 = blockIdx.y * 128 + wn * 64;

  floatx8 c[4][4] = {};

  for (int k0 = 0; k0 < D_DIM; k0 += 32) {
    bf16x16 a[4], b[4];
#pragma unroll
    for (int mi = 0; mi < 4; ++mi) {
      const int r = m0 + mi * 16 + ln;
      if (A_BF) a[mi] = frag_a_bf16(Ab + (size_t)r * D_DIM + k0, half);
      else      a[mi] = frag_a_f32(Af + (size_t)r * D_DIM + k0, half);
    }
#pragma unroll
    for (int ni = 0; ni < 4; ++ni) {
      const int n = n0 + ni * 16 + ln;
      if (W_BF) b[ni] = frag_b_bf16(Wb + (size_t)n * D_DIM + k0, half);
      else      b[ni] = frag_b_f32(Wf + (size_t)n * D_DIM + k0, half);
    }
#pragma unroll
    for (int mi = 0; mi < 4; ++mi)
#pragma unroll
      for (int ni = 0; ni < 4; ++ni)
        c[mi][ni] = wmma_bf16(a[mi], b[ni], c[mi][ni]);
  }

  // Epilogue: D rows live at half*8+vg, cols at lane&15.
#pragma unroll
  for (int mi = 0; mi < 4; ++mi) {
#pragma unroll
    for (int ni = 0; ni < 4; ++ni) {
      const int e = n0 + ni * 16 + ln;
#pragma unroll
      for (int vg = 0; vg < 8; ++vg) {
        const int r = m0 + mi * 16 + half * 8 + vg;
        const float v = c[mi][ni][vg];
        if (OUT_MODE == 2) {
          ((float*)outp)[(size_t)r * D_DIM + e] = v;
        } else {
          const int bb = r >> 11, s = r & (S_DIM - 1);
          const int h = e >> 6,  d = e & (DK_DIM - 1);
          size_t idx;
          if (OUT_MODE == 0)
            idx = (((size_t)bb * H_DIM + h) * S_DIM + s) * DK_DIM + d;
          else
            idx = (((size_t)bb * H_DIM + h) * DK_DIM + d) * S_DIM + s;
          ((__bf16*)outp)[idx] = (__bf16)v;
        }
      }
    }
  }
}

// Host-side dispatch to the right instantiation (branch-free device loops).
template <int OUT_MODE>
static void launch_gemm(bool a_bf, bool w_bf,
                        const float* Af, const __bf16* Ab,
                        const float* Wf, const __bf16* Wb,
                        void* outp, hipStream_t stream) {
  const dim3 g(32, 8, 1), b(256, 1, 1);
  if (a_bf && w_bf)
    gemm_wmma_bf16<OUT_MODE, true,  true ><<<g, b, 0, stream>>>(Af, Ab, Wf, Wb, outp);
  else if (a_bf)
    gemm_wmma_bf16<OUT_MODE, true,  false><<<g, b, 0, stream>>>(Af, Ab, Wf, Wb, outp);
  else if (w_bf)
    gemm_wmma_bf16<OUT_MODE, false, true ><<<g, b, 0, stream>>>(Af, Ab, Wf, Wb, outp);
  else
    gemm_wmma_bf16<OUT_MODE, false, false><<<g, b, 0, stream>>>(Af, Ab, Wf, Wb, outp);
}

// ---------------------------------------------------------------------------
// Flash attention, transposed-score formulation. One wave per 16-row Q tile,
// 32 keys per step. No LDS at all.
//
// S^T tile (16 keys x 16 q) D-layout: lane = 16*half + q, VGPR vg = key
// (half*8+vg within subtile). Two subtiles T0 (keys j..j+15), T1 (+16) give,
// per lane, probabilities for keys {half*8+vg} and {16+half*8+vg} -- which is
// EXACTLY the 16x32 A-fragment layout needed for P.V. Softmax reductions are
// register reductions + one shfl_xor(16); alpha/l are per-lane scalars.
// Qh,Kh: bf16 [B,H,S,64]; Vt: bf16 [B,H,64,S]; ctx: bf16 [B,S,1024].
// ---------------------------------------------------------------------------
__global__ __launch_bounds__(128) void attn_flash_wmma(
    const __bf16* __restrict__ Qh, const __bf16* __restrict__ Kh,
    const __bf16* __restrict__ Vt, const int* __restrict__ mask,
    __bf16* __restrict__ ctx) {
  const int lane = threadIdx.x & 31;
  const int wv   = threadIdx.x >> 5;
  const int half = lane >> 4;
  const int ln   = lane & 15;

  const int wid = blockIdx.x * 4 + wv;        // 0 .. B*H*(S/16)-1 = 8191
  const int b   = wid >> 11;                  // / (H * S/16)
  const int rem = wid & 2047;
  const int h   = rem >> 7;                   // / (S/16)
  const int qt  = rem & 127;

  const size_t bh = (size_t)b * H_DIM + h;
  // Q as B-fragments: B[k=d][n=q] = Q[q][d]; row n=q contiguous in d.
  const __bf16* Qp = Qh + (bh * S_DIM + qt * 16 + ln) * DK_DIM;
  const bf16x16 qb0 = frag_b_bf16(Qp, half);        // d = 0..31
  const bf16x16 qb1 = frag_b_bf16(Qp + 32, half);   // d = 32..63

  const __bf16* Kbase = Kh + bh * S_DIM * DK_DIM;
  const __bf16* Vbase = Vt + bh * DK_DIM * S_DIM;
  const int* mp = mask + (size_t)b * S_DIM;

  float mrun = -1e30f;   // running max for q = lane&15 (replicated halves)
  float lrun = 0.f;      // running sum
  floatx8 O[4] = {};     // O[m=q rows half*8+vg][n=d f*16+ln]

  const float scale = 0.125f;  // 1/sqrt(64)

  for (int j = 0; j < S_DIM; j += 32) {
    // ---- S^T tiles: T = K_tile x Q^T (A = K rows, B = Q rows) ------------
    floatx8 t0 = {}, t1 = {};
    {
      const __bf16* k0p = Kbase + (size_t)(j + ln) * DK_DIM;
      t0 = wmma_bf16(frag_a_bf16(k0p, half), qb0, t0);
      t0 = wmma_bf16(frag_a_bf16(k0p + 32, half), qb1, t0);
      const __bf16* k1p = Kbase + (size_t)(j + 16 + ln) * DK_DIM;
      t1 = wmma_bf16(frag_a_bf16(k1p, half), qb0, t1);
      t1 = wmma_bf16(frag_a_bf16(k1p + 32, half), qb1, t1);
    }

    // ---- mask by key (= VGPR row): aligned int4 loads --------------------
    const int4 ma = *(const int4*)(mp + j + half * 8);
    const int4 mb = *(const int4*)(mp + j + half * 8 + 4);
    const int4 mc = *(const int4*)(mp + j + 16 + half * 8);
    const int4 md = *(const int4*)(mp + j + 16 + half * 8 + 4);
    const int mk0[8] = {ma.x, ma.y, ma.z, ma.w, mb.x, mb.y, mb.z, mb.w};
    const int mk1[8] = {mc.x, mc.y, mc.z, mc.w, md.x, md.y, md.z, md.w};

    float s0[8], s1[8];
    float nm = mrun;
#pragma unroll
    for (int vg = 0; vg < 8; ++vg) {
      s0[vg] = mk0[vg] ? t0[vg] * scale : NEG_INF;
      s1[vg] = mk1[vg] ? t1[vg] * scale : NEG_INF;
      nm = fmaxf(nm, fmaxf(s0[vg], s1[vg]));
    }
    // merge the two key-halves (lanes q and q+16 hold keys 0-7 / 8-15)
    nm = fmaxf(nm, __shfl_xor(nm, 16, 32));

    const float alpha = __expf(mrun - nm);
    float p0[8], p1[8], ls = 0.f;
#pragma unroll
    for (int vg = 0; vg < 8; ++vg) {
      p0[vg] = __expf(s0[vg] - nm);
      p1[vg] = __expf(s1[vg] - nm);
      ls += p0[vg] + p1[vg];
    }
    ls += __shfl_xor(ls, 16, 32);
    lrun = lrun * alpha + ls;
    mrun = nm;

    // ---- broadcast alpha (per-q lane scalar) to O row dimension ----------
    float arow[8];
#pragma unroll
    for (int vg = 0; vg < 8; ++vg) arow[vg] = __shfl(alpha, half * 8 + vg, 32);
#pragma unroll
    for (int f = 0; f < 4; ++f)
#pragma unroll
      for (int vg = 0; vg < 8; ++vg) O[f][vg] *= arow[vg];

    // ---- P already in A-fragment layout: pure register pack --------------
    bf16x16 pa;
#pragma unroll
    for (int vg = 0; vg < 8; ++vg) {
      pa[vg]     = (__bf16)p0[vg];
      pa[8 + vg] = (__bf16)p1[vg];
    }

    // ---- O[16x64] += P[16x32] * V[32x64]  (Vt gives contiguous B cols) ---
#pragma unroll
    for (int f = 0; f < 4; ++f) {
      const __bf16* vp = Vbase + (size_t)(f * 16 + ln) * S_DIM + j;
      O[f] = wmma_bf16(pa, frag_b_bf16(vp, half), O[f]);
    }
  }

  // ---- epilogue: broadcast 1/l to rows, normalize, store ctx -------------
  const float inv = 1.0f / lrun;
  float irow[8];
#pragma unroll
  for (int vg = 0; vg < 8; ++vg) irow[vg] = __shfl(inv, half * 8 + vg, 32);
#pragma unroll
  for (int f = 0; f < 4; ++f) {
    const int d = f * 16 + ln;
#pragma unroll
    for (int vg = 0; vg < 8; ++vg) {
      const int s = qt * 16 + half * 8 + vg;
      ctx[((size_t)b * S_DIM + s) * D_DIM + h * DK_DIM + d] =
          (__bf16)(O[f][vg] * irow[vg]);
    }
  }
}

// ---------------------------------------------------------------------------
extern "C" void kernel_launch(void* const* d_in, const int* in_sizes, int n_in,
                              void* d_out, int out_size, void* d_ws, size_t ws_size,
                              hipStream_t stream) {
  (void)in_sizes; (void)n_in; (void)out_size;
  const float* query = (const float*)d_in[0];
  const float* key   = (const float*)d_in[1];
  const float* value = (const float*)d_in[2];
  const int*   mask  = (const int*)d_in[3];
  const float* Wq    = (const float*)d_in[4];
  const float* Wk    = (const float*)d_in[5];
  const float* Wv    = (const float*)d_in[6];
  const float* Wo    = (const float*)d_in[7];

  const size_t n_act = (size_t)B_DIM * S_DIM * D_DIM;  // 8,388,608
  const size_t n_w   = (size_t)D_DIM * D_DIM;          // 1,048,576

  // Workspace layout (bf16 elements):
  //   [0      , 4*n_act)          Qh, Kh, Vt, ctx          (64 MB, required)
  //   [4*n_act, 4*n_act+4*n_w)    Wq/Wk/Wv/Wo bf16         (+8 MB, optional)
  //   [...    , ... + 3*n_act)    query/key/value bf16     (+48 MB, optional)
  __bf16* Qh  = (__bf16*)d_ws;
  __bf16* Kh  = Qh + n_act;
  __bf16* Vt  = Kh + n_act;
  __bf16* ctx = Vt + n_act;
  __bf16* Wqb = ctx + n_act;
  __bf16* Wkb = Wqb + n_w;
  __bf16* Wvb = Wkb + n_w;
  __bf16* Wob = Wvb + n_w;
  __bf16* Xq  = Wob + n_w;
  __bf16* Xk  = Xq + n_act;
  __bf16* Xv  = Xk + n_act;

  const bool use_wbf = ws_size >= (4 * n_act + 4 * n_w) * sizeof(__bf16);
  const bool use_xbf = ws_size >= (7 * n_act + 4 * n_w) * sizeof(__bf16);

  const dim3 cblk(256, 1, 1);
  if (use_wbf) {
    cvt_f32_to_bf16<<<dim3(256), cblk, 0, stream>>>(Wq, Wqb, (int)(n_w / 4));
    cvt_f32_to_bf16<<<dim3(256), cblk, 0, stream>>>(Wk, Wkb, (int)(n_w / 4));
    cvt_f32_to_bf16<<<dim3(256), cblk, 0, stream>>>(Wv, Wvb, (int)(n_w / 4));
    cvt_f32_to_bf16<<<dim3(256), cblk, 0, stream>>>(Wo, Wob, (int)(n_w / 4));
  }
  if (use_xbf) {
    cvt_f32_to_bf16<<<dim3(1024), cblk, 0, stream>>>(query, Xq, (int)(n_act / 4));
    cvt_f32_to_bf16<<<dim3(1024), cblk, 0, stream>>>(key,   Xk, (int)(n_act / 4));
    cvt_f32_to_bf16<<<dim3(1024), cblk, 0, stream>>>(value, Xv, (int)(n_act / 4));
  }

  launch_gemm<0>(use_xbf, use_wbf, query, Xq, Wq, Wqb, (void*)Qh, stream);
  launch_gemm<0>(use_xbf, use_wbf, key,   Xk, Wk, Wkb, (void*)Kh, stream);
  launch_gemm<1>(use_xbf, use_wbf, value, Xv, Wv, Wvb, (void*)Vt, stream);

  attn_flash_wmma<<<dim3(2048, 1, 1), dim3(128, 1, 1), 0, stream>>>(
      Qh, Kh, Vt, mask, ctx);

  launch_gemm<2>(true, use_wbf, nullptr, ctx, Wo, Wob, d_out, stream);
}
